// PalaceYiJingBlock_30666066494180
// MI455X (gfx1250) — matrix operand order repaired and checked by reading, not compiled, not run on hardware
//
#include <hip/hip_runtime.h>
#include <hip/hip_bf16.h>
#include <math.h>

// ---------------------------------------------------------------------------
// PalaceYiJingBlock on MI455X (gfx1250): f16 WMMA for all matrix work,
// TDM (tensor_load_to_lds) double-buffered staging for GEMM, attention and
// GAT aggregation; f32 VALU for LN / softmax statistics. Wave32.
// ---------------------------------------------------------------------------

typedef __attribute__((ext_vector_type(16))) _Float16 v16h;
typedef __attribute__((ext_vector_type(8)))  _Float16 v8h;
typedef __attribute__((ext_vector_type(8)))  float    v8f;
typedef __attribute__((ext_vector_type(4)))  unsigned int u32x4;
typedef __attribute__((ext_vector_type(8)))  unsigned int u32x8;

#define BB   2
#define TT   1024
#define DD   1024
#define HHN  16
#define DHH  64
#define DFF  4096
#define ROWS (BB * TT)   // 2048

// ---------------------------------------------------------------- WMMA utils

__device__ __forceinline__ v8f wmma16(v16h a, v16h b, v8f c) {
  return __builtin_amdgcn_wmma_f32_16x16x32_f16(
      false, a, false, b, (short)0, c, false, false);
}

// A fragment: 16x32 f16, row-major source with leading dim lda (elements).
__device__ __forceinline__ v16h load_a_frag(const _Float16* base, int lda, int lane) {
  int r    = lane & 15;
  int koff = (lane & 16) ? 8 : 0;
  const _Float16* p = base + (size_t)r * lda + koff;
  union { v16h w; v8h h[2]; } u;
  u.h[0] = *(const v8h*)(p);
  u.h[1] = *(const v8h*)(p + 16);
  return u.w;
}

// B fragment: 32x16 f16 (KxN), source is pre-transposed Bt [N x K] row-major.
__device__ __forceinline__ v16h load_b_frag(const _Float16* base, int ldb, int lane) {
  int n    = lane & 15;
  int koff = (lane & 16) ? 8 : 0;
  const _Float16* p = base + (size_t)n * ldb + koff;
  union { v16h w; v8h h[2]; } u;
  u.h[0] = *(const v8h*)(p);
  u.h[1] = *(const v8h*)(p + 16);
  return u.w;
}

// ------------------------------------------------------- TDM 2D tile -> LDS
// Builds a D# per CDNA5 ISA ch.8 and issues TENSOR_LOAD_TO_LDS (2 SGPR groups,
// 2D tensor). dims/strides in elements of 2 bytes (data_size code 1).
__device__ __forceinline__ void tdm_load_2d(unsigned lds_off, const _Float16* gptr,
                                            unsigned tensor_d0, unsigned tensor_d1,
                                            unsigned tile_d0, unsigned tile_d1,
                                            unsigned stride0) {
  unsigned long long ga = (unsigned long long)(size_t)gptr;
  u32x4 g0;
  g0[0] = 1u;                                        // count=1 user descriptor
  g0[1] = lds_off;                                   // lds_addr (bytes)
  g0[2] = (unsigned)ga;                              // global_addr[31:0]
  g0[3] = (unsigned)((ga >> 32) & 0x01FFFFFFull)     // global_addr[56:32]
          | 0x80000000u;                             // type=2 (bits 127:126)
  u32x8 g1;
  g1[0] = 1u << 16;                                  // data_size=1 (2B), mask=0
  g1[1] = (tensor_d0 & 0xFFFFu) << 16;               // tensor_dim0[15:0] @63:48
  g1[2] = (tensor_d0 >> 16) | ((tensor_d1 & 0xFFFFu) << 16);
  g1[3] = (tensor_d1 >> 16) | (tile_d0 << 16);       // tile_dim0 @127:112
  g1[4] = tile_d1 & 0xFFFFu;                         // tile_dim1 (tile_dim2=0)
  g1[5] = stride0;                                   // tensor_dim0_stride lo
  g1[6] = 0u;
  g1[7] = 0u;
  asm volatile("tensor_load_to_lds %0, %1" :: "s"(g0), "s"(g1) : "memory");
}

// ------------------------------------------------------------- block reduces

__device__ __forceinline__ float blk_sum(float v, float* s) {
  int t = threadIdx.x;
  s[t] = v; __syncthreads();
  #pragma unroll
  for (int o = 128; o > 0; o >>= 1) {
    if (t < o) s[t] += s[t + o];
    __syncthreads();
  }
  float r = s[0]; __syncthreads();
  return r;
}

__device__ __forceinline__ float blk_max(float v, float* s) {
  int t = threadIdx.x;
  s[t] = v; __syncthreads();
  #pragma unroll
  for (int o = 128; o > 0; o >>= 1) {
    if (t < o) s[t] = fmaxf(s[t], s[t + o]);
    __syncthreads();
  }
  float r = s[0]; __syncthreads();
  return r;
}

// ----------------------------------------------------- weight cast+transpose
__global__ __launch_bounds__(256) void k_castw(const float* __restrict__ W,
                                               _Float16* __restrict__ Wt,
                                               int K, int N) {
  size_t i = (size_t)blockIdx.x * 256 + threadIdx.x;
  int k = (int)(i % (size_t)K);
  int n = (int)(i / (size_t)K);
  Wt[i] = (_Float16)W[(size_t)k * N + n];
}

// ----------------------------------------------------------- fused double LN
__global__ __launch_bounds__(256) void k_ln2(const float* __restrict__ x,
    const float* __restrict__ g1, const float* __restrict__ b1,
    const float* __restrict__ g2, const float* __restrict__ b2,
    _Float16* __restrict__ h1, _Float16* __restrict__ h2) {
  __shared__ float sbuf[256];
  int row = blockIdx.x;
  int tid = threadIdx.x;
  size_t base = (size_t)row * DD;
  float lv[4], s = 0.f;
  #pragma unroll
  for (int i = 0; i < 4; ++i) { lv[i] = x[base + tid + i * 256]; s += lv[i]; }
  float mean = blk_sum(s, sbuf) * (1.f / DD);
  float vs = 0.f;
  #pragma unroll
  for (int i = 0; i < 4; ++i) { float d = lv[i] - mean; vs += d * d; }
  float var  = blk_sum(vs, sbuf) * (1.f / DD);
  float rinv = rsqrtf(var + 1e-5f);
  #pragma unroll
  for (int i = 0; i < 4; ++i) {
    int c = tid + i * 256;
    float xn = (lv[i] - mean) * rinv;
    h1[base + c] = (_Float16)(xn * g1[c] + b1[c]);
    h2[base + c] = (_Float16)(xn * g2[c] + b2[c]);
  }
}

// ------------------------------------------- TDM-pipelined generic WMMA GEMM
// 256 thr = 8 waves; wave tile 16x64; block tile 64x128; 24 KB LDS.
__global__ __launch_bounds__(256) void k_gemm(
    const _Float16* __restrict__ A, const _Float16* __restrict__ Bt,
    const float* __restrict__ bias, const float* __restrict__ res,
    float* __restrict__ outF, _Float16* __restrict__ outH,
    int M, int N, int K, int act) {
  __shared__ _Float16 sA[2][64 * 32];
  __shared__ _Float16 sB[2][128 * 32];
  const int lane = threadIdx.x & 31;
  const int w    = threadIdx.x >> 5;
  const int wm   = w & 3;
  const int wn   = w >> 2;
  const int row0 = blockIdx.x * 64;
  const int col0 = blockIdx.y * 128;
  const int nk   = K >> 5;

  v8f acc[4];
  #pragma unroll
  for (int j = 0; j < 4; ++j)
    #pragma unroll
    for (int e = 0; e < 8; ++e) acc[j][e] = 0.f;

  if (w == 0) {
    tdm_load_2d((unsigned)(size_t)&sA[0][0], A + (size_t)row0 * K,
                (unsigned)K, (unsigned)M, 32u, 64u, (unsigned)K);
    tdm_load_2d((unsigned)(size_t)&sB[0][0], Bt + (size_t)col0 * K,
                (unsigned)K, (unsigned)N, 32u, 128u, (unsigned)K);
  }

  for (int kt = 0; kt < nk; ++kt) {
    const int cur = kt & 1;
    if (w == 0) {
      if (kt + 1 < nk) {
        const int k0n = (kt + 1) << 5;
        tdm_load_2d((unsigned)(size_t)&sA[cur ^ 1][0],
                    A + (size_t)row0 * K + k0n,
                    (unsigned)K, (unsigned)M, 32u, 64u, (unsigned)K);
        tdm_load_2d((unsigned)(size_t)&sB[cur ^ 1][0],
                    Bt + (size_t)col0 * K + k0n,
                    (unsigned)K, (unsigned)N, 32u, 128u, (unsigned)K);
        __builtin_amdgcn_s_wait_tensorcnt(2);   // stage kt landed (in-order)
      } else {
        __builtin_amdgcn_s_wait_tensorcnt(0);
      }
    }
    __syncthreads();
    v16h af = load_a_frag(&sA[cur][wm * 16 * 32], 32, lane);
    #pragma unroll
    for (int j = 0; j < 4; ++j) {
      v16h bf = load_b_frag(&sB[cur][(wn * 64 + j * 16) * 32], 32, lane);
      acc[j] = wmma16(af, bf, acc[j]);
    }
    __syncthreads();
  }

  const int nl  = lane & 15;
  const int hi8 = (lane & 16) ? 8 : 0;
  #pragma unroll
  for (int j = 0; j < 4; ++j) {
    #pragma unroll
    for (int r = 0; r < 8; ++r) {
      int mrow = row0 + wm * 16 + r + hi8;
      int ncol = col0 + wn * 64 + j * 16 + nl;
      float v = acc[j][r];
      if (bias) v += bias[ncol];
      if (act == 1) v = 0.5f * v * (1.0f + erff(v * 0.70710678118654752f)); // GELU
      size_t oi = (size_t)mrow * N + ncol;
      if (res)  v += res[oi];
      if (outF) outF[oi] = v;
      if (outH) outH[oi] = (_Float16)v;
    }
  }
}

// ------------------------------------------------------ transpose V and hw
__global__ __launch_bounds__(256) void k_tv(const _Float16* __restrict__ v,
                                            _Float16* __restrict__ vt) {
  size_t i = (size_t)blockIdx.x * 256 + threadIdx.x;
  int d = (int)(i & 63);
  int h = (int)((i >> 6) & 15);
  int t = (int)((i >> 10) & 1023);
  int b = (int)(i >> 20);
  vt[((((size_t)b * HHN + h) * DHH + d) * TT) + t] = v[i];
}

__global__ __launch_bounds__(256) void k_thw(const _Float16* __restrict__ hw,
                                             _Float16* __restrict__ hwt) {
  size_t i = (size_t)blockIdx.x * 256 + threadIdx.x;
  int d = (int)(i & 1023);
  int t = (int)((i >> 10) & 1023);
  int b = (int)(i >> 20);
  hwt[((size_t)b * DD + d) * TT + t] = hw[i];
}

// ------------------------------- palace flash attention (TDM + WMMA)
// 128 thr = 4 waves sharing one (b,h); K tile (32x64) and V tile (64x32)
// TDM-staged per 32-key stage, double-buffered; each wave owns a 16-row tile
// with online softmax; P re-fragmented through a private LDS tile.
__global__ __launch_bounds__(128) void k_attn(
    const _Float16* __restrict__ q, const _Float16* __restrict__ k,
    const _Float16* __restrict__ vt, _Float16* __restrict__ ao,
    const float* __restrict__ inter_w) {
  __shared__ _Float16 sK[2][32 * 64];    // keys s(rows) x dh  4 KB x2
  __shared__ _Float16 sV[2][64 * 32];    // vt   d(rows) x s   4 KB x2
  __shared__ _Float16 pbuf[4][16 * 32];  // per-wave P tile    4 KB
  const int lane = threadIdx.x & 31;
  const int w    = threadIdx.x >> 5;
  const int bh   = blockIdx.y;              // b*H + h
  const int b    = bh >> 4;
  const int h    = bh & 15;
  const int row0 = blockIdx.x * 64 + w * 16;
  const int nl   = lane & 15;
  const int hi8  = (lane & 16) ? 8 : 0;
  const float sig = 1.f / (1.f + __expf(-inter_w[0]));

  const _Float16* qb  = q + ((size_t)(b * TT + row0)) * DD + h * DHH;
  const _Float16* kb  = k + ((size_t)b * TT) * DD + h * DHH;
  const _Float16* vtb = vt + (size_t)bh * DHH * TT;
  _Float16* pb = &pbuf[w][0];

  v16h qa0 = load_a_frag(qb, DD, lane);        // dh chunk 0..31
  v16h qa1 = load_a_frag(qb + 32, DD, lane);   // dh chunk 32..63

  v8f o[4];
  #pragma unroll
  for (int j = 0; j < 4; ++j)
    #pragma unroll
    for (int e = 0; e < 8; ++e) o[j][e] = 0.f;

  float mrow[8], lrow[8];
  int palt[8];
  #pragma unroll
  for (int r = 0; r < 8; ++r) {
    mrow[r] = -1e30f; lrow[r] = 0.f;
    palt[r] = ((row0 + r + hi8) & 63) >> 3;
  }

  if (w == 0) {   // prologue: stage 0
    tdm_load_2d((unsigned)(size_t)&sK[0][0], kb,  DD, TT, 64u, 32u, DD);
    tdm_load_2d((unsigned)(size_t)&sV[0][0], vtb, TT, DHH, 32u, 64u, TT);
  }

  const int nst = TT / 32;
  for (int kt = 0; kt < nst; ++kt) {
    const int cur = kt & 1;
    const int s0  = kt * 32;
    if (w == 0) {
      if (kt + 1 < nst) {
        tdm_load_2d((unsigned)(size_t)&sK[cur ^ 1][0],
                    kb + (size_t)(s0 + 32) * DD, DD, TT, 64u, 32u, DD);
        tdm_load_2d((unsigned)(size_t)&sV[cur ^ 1][0],
                    vtb + (s0 + 32), TT, DHH, 32u, 64u, TT);
        __builtin_amdgcn_s_wait_tensorcnt(2);
      } else {
        __builtin_amdgcn_s_wait_tensorcnt(0);
      }
    }
    __syncthreads();                            // K/V stage kt visible
    // ---- S = (Q K^T) * scale ----
    v8f sacc[2];
    #pragma unroll
    for (int j = 0; j < 2; ++j) {
      #pragma unroll
      for (int e = 0; e < 8; ++e) sacc[j][e] = 0.f;
      const _Float16* kl = &sK[cur][(j * 16) * 64];
      sacc[j] = wmma16(qa0, load_b_frag(kl,      64, lane), sacc[j]);
      sacc[j] = wmma16(qa1, load_b_frag(kl + 32, 64, lane), sacc[j]);
    }
    // ---- online softmax + post-softmax palace mask ----
    #pragma unroll
    for (int r = 0; r < 8; ++r) {
      float a0 = sacc[0][r] * 0.125f;
      float a1 = sacc[1][r] * 0.125f;
      float mx = fmaxf(a0, a1);
      mx = fmaxf(mx, __shfl_xor(mx, 1, 32));
      mx = fmaxf(mx, __shfl_xor(mx, 2, 32));
      mx = fmaxf(mx, __shfl_xor(mx, 4, 32));
      mx = fmaxf(mx, __shfl_xor(mx, 8, 32));
      float mn   = fmaxf(mrow[r], mx);
      float corr = __expf(mrow[r] - mn);
      mrow[r] = mn;
      lrow[r] *= corr;
      #pragma unroll
      for (int j = 0; j < 4; ++j) o[j][r] = o[j][r] * corr;
      float p0 = __expf(a0 - mn);
      float p1 = __expf(a1 - mn);
      float ps = p0 + p1;
      ps += __shfl_xor(ps, 1, 32);
      ps += __shfl_xor(ps, 2, 32);
      ps += __shfl_xor(ps, 4, 32);
      ps += __shfl_xor(ps, 8, 32);
      lrow[r] += ps;
      int c0 = s0 + nl, c1 = s0 + 16 + nl;
      float w0 = (((c0 & 63) >> 3) == palt[r]) ? 1.f : sig;
      float w1 = (((c1 & 63) >> 3) == palt[r]) ? 1.f : sig;
      pb[(r + hi8) * 32 + nl]      = (_Float16)(p0 * w0);
      pb[(r + hi8) * 32 + 16 + nl] = (_Float16)(p1 * w1);
    }
    asm volatile("" ::: "memory");   // LDS in-order within a wave (DScnt)
    v16h pa = load_a_frag(pb, 32, lane);
    #pragma unroll
    for (int j = 0; j < 4; ++j) {
      v16h vb = load_b_frag(&sV[cur][(j * 16) * 32], 32, lane);
      o[j] = wmma16(pa, vb, o[j]);
    }
    __syncthreads();                            // stage kt fully consumed
  }

  _Float16* ob = ao + ((size_t)(b * TT + row0)) * DD + h * DHH;
  #pragma unroll
  for (int j = 0; j < 4; ++j)
    #pragma unroll
    for (int r = 0; r < 8; ++r) {
      float val = o[j][r] / lrow[r];
      ob[(size_t)(r + hi8) * DD + j * 16 + nl] = (_Float16)val;
    }
}

// ---------------------------------------------------------------- GAT pieces
__global__ __launch_bounds__(256) void k_sctc(const _Float16* __restrict__ hw,
    const float* __restrict__ a_src, const float* __restrict__ a_dst,
    float* __restrict__ sc, float* __restrict__ tc) {
  __shared__ float sbuf[256];
  int row = blockIdx.x;
  int tid = threadIdx.x;
  const _Float16* hr = hw + (size_t)row * DD;
  float s1 = 0.f, s2 = 0.f;
  #pragma unroll
  for (int i = 0; i < 4; ++i) {
    int c = tid + i * 256;
    float v = (float)hr[c];
    s1 += v * a_src[c];
    s2 += v * a_dst[c];
  }
  float r1 = blk_sum(s1, sbuf);
  float r2 = blk_sum(s2, sbuf);
  if (tid == 0) { sc[row] = r1; tc[row] = r2; }
}

__global__ __launch_bounds__(256) void k_gatstats(const float* __restrict__ sc,
    const float* __restrict__ tc, float* __restrict__ gm, float* __restrict__ gl) {
  __shared__ float sbuf[256];
  int t = blockIdx.x, b = blockIdx.y;
  int tid = threadIdx.x;
  float sct = sc[b * TT + t];
  const float* tcb = tc + b * TT;
  float mx = -1e30f;
  for (int s = tid; s < TT; s += 256) {
    float e = sct + tcb[s];
    e = e > 0.f ? e : 0.2f * e;
    mx = fmaxf(mx, e);
  }
  float m = blk_max(mx, sbuf);
  float su = 0.f;
  for (int s = tid; s < TT; s += 256) {
    float e = sct + tcb[s];
    e = e > 0.f ? e : 0.2f * e;
    su += __expf(e - m);
  }
  float l = blk_sum(su, sbuf);
  if (tid == 0) { gm[b * TT + t] = m; gl[b * TT + t] = l; }
}

// h_gat = softmax(e) @ hw : A-fragment synthesized analytically in registers,
// hw^T tile (128x32 f16) TDM-staged per stage and shared by all 8 waves.
__global__ __launch_bounds__(256) void k_gat(const _Float16* __restrict__ hwt,
    const float* __restrict__ sc, const float* __restrict__ tc,
    const float* __restrict__ gm, const float* __restrict__ gl,
    float* __restrict__ hgat) {
  __shared__ _Float16 sB[2][128 * 32];   // 8 KB x2
  const int lane = threadIdx.x & 31;
  const int w    = threadIdx.x >> 5;
  const int wm = w & 3, wn = w >> 2;
  const int b    = blockIdx.z;
  const int row0 = blockIdx.x * 64 + wm * 16;
  const int col0 = blockIdx.y * 128;          // block column base
  const int rl   = lane & 15;
  const int trow = row0 + rl;
  const int koff = (lane & 16) ? 8 : 0;

  const float scl  = sc[b * TT + trow];
  const float ml   = gm[b * TT + trow];
  const float invl = 1.f / gl[b * TT + trow];
  const float* tcb = tc + b * TT;
  const _Float16* hb = hwt + (size_t)b * DD * TT + (size_t)col0 * TT;

  v8f acc[4];
  #pragma unroll
  for (int j = 0; j < 4; ++j)
    #pragma unroll
    for (int e = 0; e < 8; ++e) acc[j][e] = 0.f;

  if (w == 0)   // prologue
    tdm_load_2d((unsigned)(size_t)&sB[0][0], hb, TT, DD, 32u, 128u, TT);

  const int nst = TT / 32;
  for (int kt = 0; kt < nst; ++kt) {
    const int cur = kt & 1;
    const int s0  = kt * 32;
    if (w == 0) {
      if (kt + 1 < nst) {
        tdm_load_2d((unsigned)(size_t)&sB[cur ^ 1][0],
                    hb + (s0 + 32), TT, DD, 32u, 128u, TT);
        __builtin_amdgcn_s_wait_tensorcnt(1);
      } else {
        __builtin_amdgcn_s_wait_tensorcnt(0);
      }
    }
    // synthesize A fragment (independent of the staged LDS tile)
    v16h af;
    #pragma unroll
    for (int e = 0; e < 8; ++e) {
      int s1 = s0 + koff + e;
      float e1 = scl + tcb[s1];
      e1 = e1 > 0.f ? e1 : 0.2f * e1;
      af[e] = (_Float16)(__expf(e1 - ml) * invl);
      int s2 = s1 + 16;
      float e2 = scl + tcb[s2];
      e2 = e2 > 0.f ? e2 : 0.2f * e2;
      af[e + 8] = (_Float16)(__expf(e2 - ml) * invl);
    }
    __syncthreads();                            // hw tile stage kt visible
    #pragma unroll
    for (int j = 0; j < 4; ++j) {
      v16h bf = load_b_frag(&sB[cur][(wn * 64 + j * 16) * 32], 32, lane);
      acc[j] = wmma16(af, bf, acc[j]);
    }
    __syncthreads();                            // stage kt fully consumed
  }

  const int nl  = lane & 15;
  const int hi8 = (lane & 16) ? 8 : 0;
  #pragma unroll
  for (int j = 0; j < 4; ++j)
    #pragma unroll
    for (int r = 0; r < 8; ++r) {
      int t = row0 + r + hi8;
      int c = col0 + wn * 64 + j * 16 + nl;
      hgat[((size_t)b * TT + t) * DD + c] = acc[j][r];
    }
}

// --------------------------------------------- gated fusion + LN3 (fused)
__global__ __launch_bounds__(256) void k_fuse(const float* __restrict__ x,
    const float* __restrict__ hp, const float* __restrict__ hg,
    const float* __restrict__ gate, const float* __restrict__ g3,
    const float* __restrict__ b3, float* __restrict__ x2,
    _Float16* __restrict__ hf) {
  __shared__ float sbuf[256];
  int row = blockIdx.x;
  int tid = threadIdx.x;
  float alpha = 1.f / (1.f + __expf(-gate[0]));
  size_t base = (size_t)row * DD;
  float lv[4], s = 0.f;
  #pragma unroll
  for (int i = 0; i < 4; ++i) {
    size_t c = base + tid + i * 256;
    float v = x[c] + alpha * hp[c] + (1.f - alpha) * hg[c];
    lv[i] = v; x2[c] = v; s += v;
  }
  float mean = blk_sum(s, sbuf) * (1.f / DD);
  float vs = 0.f;
  #pragma unroll
  for (int i = 0; i < 4; ++i) { float d = lv[i] - mean; vs += d * d; }
  float var  = blk_sum(vs, sbuf) * (1.f / DD);
  float rinv = rsqrtf(var + 1e-5f);
  #pragma unroll
  for (int i = 0; i < 4; ++i) {
    int c = tid + i * 256;
    hf[base + c] = (_Float16)((lv[i] - mean) * rinv * g3[c] + b3[c]);
  }
}

// ---------------------------------------------------------------------------

extern "C" void kernel_launch(void* const* d_in, const int* in_sizes, int n_in,
                              void* d_out, int out_size, void* d_ws, size_t ws_size,
                              hipStream_t stream) {
  (void)in_sizes; (void)n_in; (void)out_size; (void)ws_size;
  const float* x     = (const float*)d_in[0];
  const float* Wq    = (const float*)d_in[1];
  const float* bq    = (const float*)d_in[2];
  const float* Wk    = (const float*)d_in[3];
  const float* bk    = (const float*)d_in[4];
  const float* Wv    = (const float*)d_in[5];
  const float* bv    = (const float*)d_in[6];
  const float* Wo    = (const float*)d_in[7];
  const float* bo    = (const float*)d_in[8];
  const float* intw  = (const float*)d_in[9];
  const float* Wg    = (const float*)d_in[10];
  const float* a_src = (const float*)d_in[11];
  const float* a_dst = (const float*)d_in[12];
  const float* g1    = (const float*)d_in[13];
  const float* b1    = (const float*)d_in[14];
  const float* g2    = (const float*)d_in[15];
  const float* b2    = (const float*)d_in[16];
  const float* g3    = (const float*)d_in[17];
  const float* b3    = (const float*)d_in[18];
  const float* gate  = (const float*)d_in[19];
  const float* W1    = (const float*)d_in[20];
  const float* bf1   = (const float*)d_in[21];
  const float* W2    = (const float*)d_in[22];
  const float* bf2   = (const float*)d_in[23];

  char* p = (char*)d_ws;
  auto carve = [&](size_t bytes) -> void* {
    void* r = (void*)p;
    p += (bytes + 255) & ~((size_t)255);
    return r;
  };
  const size_t NBT = (size_t)ROWS * DD;            // 2M elements
  _Float16* h1h  = (_Float16*)carve(NBT * 2);
  _Float16* h2h  = (_Float16*)carve(NBT * 2);
  _Float16* WqT  = (_Float16*)carve((size_t)DD * DD * 2);
  _Float16* WkT  = (_Float16*)carve((size_t)DD * DD * 2);
  _Float16* WvT  = (_Float16*)carve((size_t)DD * DD * 2);
  _Float16* WoT  = (_Float16*)carve((size_t)DD * DD * 2);
  _Float16* WgT  = (_Float16*)carve((size_t)DD * DD * 2);
  _Float16* W1T  = (_Float16*)carve((size_t)DD * DFF * 2);
  _Float16* W2T  = (_Float16*)carve((size_t)DFF * DD * 2);
  _Float16* qh   = (_Float16*)carve(NBT * 2);
  _Float16* kh   = (_Float16*)carve(NBT * 2);
  _Float16* vh   = (_Float16*)carve(NBT * 2);
  _Float16* vth  = (_Float16*)carve(NBT * 2);
  _Float16* aoh  = (_Float16*)carve(NBT * 2);
  float*    hpal = (float*)carve(NBT * 4);
  _Float16* hwh  = (_Float16*)carve(NBT * 2);
  _Float16* hwth = (_Float16*)carve(NBT * 2);
  float*    scv  = (float*)carve((size_t)ROWS * 4);
  float*    tcv  = (float*)carve((size_t)ROWS * 4);
  float*    gmv  = (float*)carve((size_t)ROWS * 4);
  float*    glv  = (float*)carve((size_t)ROWS * 4);
  float*    hgat = (float*)carve(NBT * 4);
  float*    x2f  = (float*)carve(NBT * 4);
  _Float16* hfh  = (_Float16*)carve(NBT * 2);
  _Float16* f1h  = (_Float16*)carve((size_t)ROWS * DFF * 2);

  // 1) weights -> f16, pre-transposed [N x K]
  int nb1 = (DD * DD) / 256;
  k_castw<<<nb1, 256, 0, stream>>>(Wq, WqT, DD, DD);
  k_castw<<<nb1, 256, 0, stream>>>(Wk, WkT, DD, DD);
  k_castw<<<nb1, 256, 0, stream>>>(Wv, WvT, DD, DD);
  k_castw<<<nb1, 256, 0, stream>>>(Wo, WoT, DD, DD);
  k_castw<<<nb1, 256, 0, stream>>>(Wg, WgT, DD, DD);
  k_castw<<<(DD * DFF) / 256, 256, 0, stream>>>(W1, W1T, DD, DFF);
  k_castw<<<(DFF * DD) / 256, 256, 0, stream>>>(W2, W2T, DFF, DD);

  // 2) fused LN1/LN2
  k_ln2<<<ROWS, 256, 0, stream>>>(x, g1, b1, g2, b2, h1h, h2h);

  // 3) Q/K/V projections (TDM + WMMA)
  dim3 gD(ROWS / 64, DD / 128);
  k_gemm<<<gD, 256, 0, stream>>>(h1h, WqT, bq, nullptr, nullptr, qh, ROWS, DD, DD, 0);
  k_gemm<<<gD, 256, 0, stream>>>(h1h, WkT, bk, nullptr, nullptr, kh, ROWS, DD, DD, 0);
  k_gemm<<<gD, 256, 0, stream>>>(h1h, WvT, bv, nullptr, nullptr, vh, ROWS, DD, DD, 0);
  k_tv<<<(int)(NBT / 256), 256, 0, stream>>>(vh, vth);

  // 4) palace flash attention (TDM + WMMA) + output projection
  k_attn<<<dim3(TT / 64, BB * HHN), 128, 0, stream>>>(qh, kh, vth, aoh, intw);
  k_gemm<<<gD, 256, 0, stream>>>(aoh, WoT, bo, nullptr, hpal, nullptr, ROWS, DD, DD, 0);

  // 5) GAT branch
  k_gemm<<<gD, 256, 0, stream>>>(h2h, WgT, nullptr, nullptr, nullptr, hwh, ROWS, DD, DD, 0);
  k_thw<<<(int)(NBT / 256), 256, 0, stream>>>(hwh, hwth);
  k_sctc<<<ROWS, 256, 0, stream>>>(hwh, a_src, a_dst, scv, tcv);
  k_gatstats<<<dim3(TT, BB), 256, 0, stream>>>(scv, tcv, gmv, glv);
  k_gat<<<dim3(TT / 64, DD / 128, BB), 256, 0, stream>>>(hwth, scv, tcv, gmv, glv, hgat);

  // 6) gated fusion + LN3
  k_fuse<<<ROWS, 256, 0, stream>>>(x, hpal, hgat, gate, g3, b3, x2f, hfh);

  // 7) FFN: GELU GEMM then down-proj with residual into d_out (f32)
  k_gemm<<<dim3(ROWS / 64, DFF / 128), 256, 0, stream>>>(
      hfh, W1T, bf1, nullptr, nullptr, f1h, ROWS, DFF, DD, 1);
  k_gemm<<<dim3(ROWS / 64, DD / 128), 256, 0, stream>>>(
      f1h, W2T, bf2, x2f, (float*)d_out, nullptr, ROWS, DD, DFF, 0);
}
